// CHGNetBondGraphConv_86861418594830
// MI455X (gfx1250) — matrix operand order repaired and checked by reading, not compiled, not run on hardware
//
#include <hip/hip_runtime.h>

typedef __attribute__((ext_vector_type(16))) __bf16 v16bf;
typedef __attribute__((ext_vector_type(8)))  float  v8f;

#define N_NODES 100000
#define N_EDGES 1000000
#define DF   64
#define DIN  192
#define HID  64
#define WAVES 4

union FragB { v16bf v; uint4 q[2]; };

__device__ __forceinline__ unsigned short f2bf(float x) {
  union { float f; unsigned u; } c; c.f = x;
  return (unsigned short)((c.u + 0x7FFFu + ((c.u >> 16) & 1u)) >> 16);
}

// A fragment: 16x32 bf16 tile from an LDS row-major tile (row stride = `stride` elems).
// Lanes 0-15: M=lane, halves = [k0+0..7, k0+16..23]; lanes 16-31: M=lane-16, +8.
__device__ __forceinline__ v16bf load_a_frag(const unsigned short* tile, int row,
                                             int stride, int k0, int laneHi) {
  FragB f;
  const unsigned short* p = tile + row * stride + k0 + (laneHi ? 8 : 0);
  f.q[0] = *(const uint4*)p;
  f.q[1] = *(const uint4*)(p + 16);
  return f.v;
}

// B fragment: 32x16 bf16 from global transposed weights Wt[64][K] (row-major).
// Lanes 0-15: N=lane, halves = Wt[N][k0+0..7, k0+16..23]; lanes 16-31: N=lane-16, +8.
__device__ __forceinline__ v16bf load_b_frag(const unsigned short* wt, int ncol,
                                             int K, int k0, int laneHi) {
  FragB f;
  const unsigned short* p = wt + ncol * K + k0 + (laneHi ? 8 : 0);
  f.q[0] = *(const uint4*)p;
  f.q[1] = *(const uint4*)(p + 16);
  return f.v;
}

// acc[4] = A(16xK, LDS) @ Wt^T (K x 64) as four 16x16 f32 tiles
__device__ __forceinline__ void gemm_tile(const unsigned short* aTile, int K,
                                          const unsigned short* Wt,
                                          int lane, v8f acc[4]) {
  const int laneHi = lane >> 4;
  const int nlane  = lane & 15;
  v8f zero = {};
  #pragma unroll
  for (int n = 0; n < 4; ++n) acc[n] = zero;
  for (int k0 = 0; k0 < K; k0 += 32) {
    v16bf a = load_a_frag(aTile, nlane, K, k0, laneHi);
    #pragma unroll
    for (int n = 0; n < 4; ++n) {
      v16bf b = load_b_frag(Wt + n * 16 * K, nlane, K, k0, laneHi);
      acc[n] = __builtin_amdgcn_wmma_f32_16x16x32_bf16(
          false, a, false, b, (short)0, acc[n], false, false);
    }
  }
}

__device__ __forceinline__ void act_inplace(v8f acc[4], const float* bias,
                                            int lane, bool sigmoid_act) {
  const int nlane = lane & 15;
  #pragma unroll
  for (int n = 0; n < 4; ++n) {
    float bn = bias[n * 16 + nlane];
    #pragma unroll
    for (int r = 0; r < 8; ++r) {
      float v = acc[n][r] + bn;
      float s = 1.f / (1.f + __expf(-v));
      acc[n][r] = sigmoid_act ? s : v * s;
    }
  }
}

// Store activated 16x64 tile (C layout) to LDS row-major [16][64] bf16 (layer-2 A matrix)
__device__ __forceinline__ void store_tile(unsigned short* dst, const v8f acc[4], int lane) {
  const int laneHi = lane >> 4;
  const int nlane  = lane & 15;
  #pragma unroll
  for (int n = 0; n < 4; ++n) {
    #pragma unroll
    for (int r = 0; r < 8; ++r) {
      dst[(r + laneHi * 8) * HID + n * 16 + nlane] = f2bf(acc[n][r]);
    }
  }
}

__device__ __forceinline__ void gated_mlp_tile(
    const unsigned short* xA, unsigned short* hA, unsigned short* gA,
    const unsigned short* W1t, const unsigned short* W2t,
    const unsigned short* G1t, const unsigned short* G2t,
    const float* b1, const float* b2, const float* c1, const float* c2,
    int lane, v8f out[4]) {
  v8f acc[4];
  gemm_tile(xA, DIN, W1t, lane, acc);
  act_inplace(acc, b1, lane, false);
  store_tile(hA, acc, lane);
  gemm_tile(xA, DIN, G1t, lane, acc);
  act_inplace(acc, c1, lane, false);
  store_tile(gA, acc, lane);
  asm volatile("s_wait_dscnt 0" ::: "memory");
  v8f h2[4];
  gemm_tile(hA, HID, W2t, lane, h2);
  act_inplace(h2, b2, lane, false);
  v8f g2[4];
  gemm_tile(gA, HID, G2t, lane, g2);
  act_inplace(g2, c2, lane, true);
  #pragma unroll
  for (int n = 0; n < 4; ++n) out[n] = h2[n] * g2[n];
}

// PHASE 1: messages + atomic scatter into node output (pre-initialized with node_features)
// PHASE 2: edge update, residual write
template <int PHASE>
__global__ void __launch_bounds__(32 * WAVES) chgnet_phase(
    const float* __restrict__ gatherF,   // node_features (P1) or new node features (P2)
    const float* __restrict__ edgeF,
    const float* __restrict__ sw,
    const int* __restrict__ srcI, const int* __restrict__ dstI,
    const unsigned short* __restrict__ W1t, const unsigned short* __restrict__ W2t,
    const unsigned short* __restrict__ G1t, const unsigned short* __restrict__ G2t,
    const float* __restrict__ b1, const float* __restrict__ b2,
    const float* __restrict__ c1, const float* __restrict__ c2,
    float* __restrict__ outPtr) {
  __shared__ __align__(16) unsigned short xA[WAVES][16 * DIN];
  __shared__ __align__(16) unsigned short hA[WAVES][16 * HID];
  __shared__ __align__(16) unsigned short gA[WAVES][16 * HID];

  const int wave = threadIdx.x >> 5;
  const int lane = threadIdx.x & 31;
  const int tile = blockIdx.x * WAVES + wave;
  const int ebase = tile * 16;
  if (ebase >= N_EDGES) return;

  // ---- stage concat([nf[src], ef, nf[dst]]) -> bf16 LDS tile [16][192] ----
  #pragma unroll
  for (int it = 0; it < 24; ++it) {
    int idx = (it * 32 + lane) * 4;       // element index in 16x192 tile
    int el  = idx / DIN;
    int c   = idx % DIN;
    int e   = ebase + el;
    const float* srcp;
    if (c < 64)        srcp = gatherF + (long)srcI[e] * DF + c;
    else if (c < 128)  srcp = edgeF   + (long)e * DF + (c - 64);
    else               srcp = gatherF + (long)dstI[e] * DF + (c - 128);
    float4 v = *(const float4*)srcp;
    uint2 packed;
    packed.x = (unsigned)f2bf(v.x) | ((unsigned)f2bf(v.y) << 16);
    packed.y = (unsigned)f2bf(v.z) | ((unsigned)f2bf(v.w) << 16);
    *(uint2*)&xA[wave][el * DIN + c] = packed;
  }
  asm volatile("s_wait_dscnt 0" ::: "memory");

  v8f out[4];
  gated_mlp_tile(xA[wave], hA[wave], gA[wave],
                 W1t, W2t, G1t, G2t, b1, b2, c1, c2, lane, out);

  const int laneHi = lane >> 4;
  const int nlane  = lane & 15;

  if constexpr (PHASE == 1) {
    int se[8], de[8];
    #pragma unroll
    for (int r = 0; r < 8; ++r) {
      int e = ebase + r + laneHi * 8;
      se[r] = srcI[e];
      de[r] = dstI[e];
    }
    #pragma unroll
    for (int n = 0; n < 4; ++n) {
      int feat = n * 16 + nlane;
      #pragma unroll
      for (int r = 0; r < 8; ++r) {
        float m = out[n][r];
        m *= sw[(long)se[r] * DF + feat] * sw[(long)de[r] * DF + feat];
        __hip_atomic_fetch_add(outPtr + (long)de[r] * DF + feat, m,
                               __ATOMIC_RELAXED, __HIP_MEMORY_SCOPE_AGENT);
      }
    }
  } else {
    (void)sw;
    #pragma unroll
    for (int n = 0; n < 4; ++n) {
      int feat = n * 16 + nlane;
      #pragma unroll
      for (int r = 0; r < 8; ++r) {
        long e = ebase + r + laneHi * 8;
        outPtr[e * DF + feat] = edgeF[e * DF + feat] + out[n][r];
      }
    }
  }
}

__global__ void copy_node_kernel(const float* __restrict__ nf, float* __restrict__ out, int n) {
  int i = blockIdx.x * blockDim.x + threadIdx.x;
  if (i < n) out[i] = nf[i];
}

// W is [K][64] row-major (jax (fan_in, fan_out)); produce Wt[64][K] bf16
__global__ void transpose_w_kernel(const float* __restrict__ W, unsigned short* __restrict__ Wt, int K) {
  int i = blockIdx.x * blockDim.x + threadIdx.x;
  if (i < 64 * K) {
    int n = i / K, k = i % K;
    Wt[i] = f2bf(W[k * 64 + n]);
  }
}

extern "C" void kernel_launch(void* const* d_in, const int* in_sizes, int n_in,
                              void* d_out, int out_size, void* d_ws, size_t ws_size,
                              hipStream_t stream) {
  (void)in_sizes; (void)n_in; (void)out_size; (void)ws_size;
  const float* nodeF = (const float*)d_in[0];
  const float* edgeF = (const float*)d_in[1];
  const float* sw    = (const float*)d_in[2];
  const float* p[16];
  for (int i = 0; i < 16; ++i) p[i] = (const float*)d_in[3 + i];
  const int* srcI = (const int*)d_in[19];
  const int* dstI = (const int*)d_in[20];

  float* outNode = (float*)d_out;
  float* outEdge = outNode + (size_t)N_NODES * DF;

  unsigned short* ws   = (unsigned short*)d_ws;
  unsigned short* nW1t = ws;                    // 64*192
  unsigned short* nG1t = nW1t + 64 * 192;
  unsigned short* nW2t = nG1t + 64 * 192;       // 64*64
  unsigned short* nG2t = nW2t + 64 * 64;
  unsigned short* eW1t = nG2t + 64 * 64;
  unsigned short* eG1t = eW1t + 64 * 192;
  unsigned short* eW2t = eG1t + 64 * 192;
  unsigned short* eG2t = eW2t + 64 * 64;

  const int t1 = (64 * 192 + 255) / 256, t2 = (64 * 64 + 255) / 256;
  // params order: W1,b1,W2,b2,G1,c1,G2,c2 (node then edge)
  transpose_w_kernel<<<t1, 256, 0, stream>>>(p[0],  nW1t, 192);
  transpose_w_kernel<<<t2, 256, 0, stream>>>(p[2],  nW2t, 64);
  transpose_w_kernel<<<t1, 256, 0, stream>>>(p[4],  nG1t, 192);
  transpose_w_kernel<<<t2, 256, 0, stream>>>(p[6],  nG2t, 64);
  transpose_w_kernel<<<t1, 256, 0, stream>>>(p[8],  eW1t, 192);
  transpose_w_kernel<<<t2, 256, 0, stream>>>(p[10], eW2t, 64);
  transpose_w_kernel<<<t1, 256, 0, stream>>>(p[12], eG1t, 192);
  transpose_w_kernel<<<t2, 256, 0, stream>>>(p[14], eG2t, 64);

  copy_node_kernel<<<(N_NODES * DF + 255) / 256, 256, 0, stream>>>(
      nodeF, outNode, N_NODES * DF);

  dim3 grid(N_EDGES / 16 / WAVES);  // 15625, exact
  chgnet_phase<1><<<grid, 32 * WAVES, 0, stream>>>(
      nodeF, edgeF, sw, srcI, dstI,
      nW1t, nW2t, nG1t, nG2t, p[1], p[3], p[5], p[7], outNode);
  chgnet_phase<2><<<grid, 32 * WAVES, 0, stream>>>(
      outNode, edgeF, sw, srcI, dstI,
      eW1t, eW2t, eG1t, eG2t, p[9], p[11], p[13], p[15], outEdge);
}
